// SimpleNetwork_19172734009897
// MI455X (gfx1250) — matrix operand order repaired and checked by reading, not compiled, not run on hardware
//
#include <hip/hip_runtime.h>
#include <hip/hip_bf16.h>

// ---------------------------------------------------------------------------
// CDNA5 (gfx1250) GNN forward:  wave32 + v_wmma_f32_16x16x32_f16 tiles.
//  - biases of the first MLP layer folded into the K-padding of the WMMA
//  - fast SiLU via v_rcp_f32 (no IEEE divide expansion)
// ---------------------------------------------------------------------------

typedef __attribute__((ext_vector_type(16))) _Float16 v16h;
typedef __attribute__((ext_vector_type(8)))  float    v8f;
typedef __attribute__((ext_vector_type(8)))  _Float16 h8v;

#define N_NODES  100000
#define N_EDGES  3200000
#define F_NODE   32
#define F_EDGE   16
#define MSG      4
#define HID      32
#define N_GRAPHS 256

#define WPB 4   // waves per block (128 threads)

__device__ __forceinline__ float silu_f(float x) {
    // x * rcp(1 + exp(-x)) : v_exp_f32 + v_rcp_f32 + 2 mul/add, no div expansion
    return x * __builtin_amdgcn_rcpf(1.0f + __expf(-x));
}

// ---------------------------------------------------------------------------
// zero fill
// ---------------------------------------------------------------------------
__global__ void zero_f32(float* __restrict__ p, int n) {
    int i = blockIdx.x * blockDim.x + threadIdx.x;
    if (i < n) p[i] = 0.0f;
}

// ---------------------------------------------------------------------------
// Edge message kernel: one wave computes 16 edges per tile.
//   m_in[16 x 80 | 1 | pad->96] @ [ew0;eb0][96x32] -> silu -> hid[16x32]
//   hid @ ew1[32x4] (pad N->16) + eb1              -> silu -> msg[16x4]
//   atomicAdd msg into sums[src]
// ---------------------------------------------------------------------------
__global__ __launch_bounds__(128)
void edge_msg_kernel(const float* __restrict__ h,    // [N,32]
                     const int*   __restrict__ eidx, // [2,E]: src=[0..E), dst=[E..2E)
                     const float* __restrict__ ea,   // [E,16]
                     const float* __restrict__ ew0,  // [80,32] layer slice
                     const float* __restrict__ eb0,  // [32]
                     const float* __restrict__ ew1,  // [32,4]
                     const float* __restrict__ eb1,  // [4]
                     float* __restrict__ sums,       // [N,4] pre-zeroed
                     int nEdges)
{
    const int lane  = threadIdx.x & 31;
    const int wave  = threadIdx.x >> 5;
    const int l16   = lane & 15;
    const int lhalf = lane >> 4;    // 0: lanes 0-15, 1: lanes 16-31

    // per-wave LDS: A-panel 16x96 f16 + hid tile 16x32 f16
    __shared__ __align__(16) _Float16 lds[WPB][16 * 96 + 16 * 32];
    _Float16* panel = &lds[wave][0];
    _Float16* hidT  = &lds[wave][16 * 96];

    // ---- load weight B fragments (contiguous K order, same as A: lane<16 -> col N=l16,
    //      K rows kb..kb+15 with kb = c*32 + lhalf*16).
    //      K row 80 carries the bias (A column 80 is the constant 1.0). ----
    v16h Bw0[3][2];
#pragma unroll
    for (int c = 0; c < 3; ++c)
#pragma unroll
        for (int nh = 0; nh < 2; ++nh) {
            const int n  = nh * 16 + l16;
            const int kb = c * 32 + lhalf * 16;
#pragma unroll
            for (int j = 0; j < 16; ++j) {
                const int k = kb + j;
                float w = 0.0f;
                if (k < 80)       w = ew0[k * 32 + n];
                else if (k == 80) w = eb0[n];
                Bw0[c][nh][j] = (_Float16)w;
            }
        }
    v16h Bw1;
    {
        const int kb = lhalf * 16;
#pragma unroll
        for (int j = 0; j < 16; ++j)
            Bw1[j] = (_Float16)((l16 < MSG) ? ew1[(kb + j) * MSG + l16] : 0.0f);
    }
    const float bias1 = (l16 < MSG) ? eb1[l16] : 0.0f;

    const int numTiles = (nEdges + 15) >> 4;
    const int gwave    = blockIdx.x * WPB + wave;
    const int wstride  = gridDim.x * WPB;

    for (int t = gwave; t < numTiles; t += wstride) {
        const int base = t << 4;
        const int cnt  = (nEdges - base < 16) ? (nEdges - base) : 16;

        // ---- stage panel: 2 lanes per edge row ----
        const int er = lane >> 1;   // edge row 0..15
        const int eh = lane & 1;    // half
        if (er < cnt) {
            const int e = base + er;
            const int s = eidx[e];
            const int d = eidx[nEdges + e];
            const float* hs = h  + (size_t)s * 32 + eh * 16;
            const float* hd = h  + (size_t)d * 32 + eh * 16;
            const float* ep = ea + (size_t)e * 16 + eh * 8;
            _Float16* row = panel + er * 96;
#pragma unroll
            for (int j = 0; j < 16; ++j) row[eh * 16 + j]      = (_Float16)hs[j];
#pragma unroll
            for (int j = 0; j < 16; ++j) row[32 + eh * 16 + j] = (_Float16)hd[j];
#pragma unroll
            for (int j = 0; j < 8;  ++j) row[64 + eh * 8 + j]  = (_Float16)ep[j];
            // pad cols 80..95; col 80 = 1.0 (bias row selector)
#pragma unroll
            for (int j = 0; j < 8;  ++j)
                row[80 + eh * 8 + j] = (_Float16)((eh == 0 && j == 0) ? 1.0f : 0.0f);
        } else {
            _Float16* row = panel + er * 96;
#pragma unroll
            for (int j = 0; j < 48; ++j) row[eh * 48 + j] = (_Float16)0.0f;
        }
        asm volatile("s_wait_dscnt 0" ::: "memory");

        // ---- GEMM1: 3 K-chunks of 32, two N tiles ----
        v8f acc0 = {}; v8f acc1 = {};
#pragma unroll
        for (int c = 0; c < 3; ++c) {
            // A layout: lane<16 row M=l16, K = c*32 + {0..15}; lane>=16: +16
            const h8v* rp = (const h8v*)(panel + l16 * 96 + c * 32 + lhalf * 16);
            const h8v lo = rp[0];
            const h8v hi = rp[1];
            v16h a;
#pragma unroll
            for (int j = 0; j < 8; ++j) { a[j] = lo[j]; a[8 + j] = hi[j]; }
            acc0 = __builtin_amdgcn_wmma_f32_16x16x32_f16(false, a, false, Bw0[c][0],
                                                          (short)0, acc0, false, false);
            acc1 = __builtin_amdgcn_wmma_f32_16x16x32_f16(false, a, false, Bw0[c][1],
                                                          (short)0, acc1, false, false);
        }

        // ---- silu (bias already folded), stage hid tile 16x32 f16 ----
#pragma unroll
        for (int v = 0; v < 8; ++v) {
            const int m = v + lhalf * 8;   // D layout: lane>=16 holds M=v+8
            hidT[m * 32 + l16]      = (_Float16)silu_f(acc0[v]);
            hidT[m * 32 + 16 + l16] = (_Float16)silu_f(acc1[v]);
        }
        asm volatile("s_wait_dscnt 0" ::: "memory");

        // ---- GEMM2: hid[16x32] @ ew1 (N padded to 16) ----
        v16h a2;
        {
            const h8v* rp = (const h8v*)(hidT + l16 * 32 + lhalf * 16);
            const h8v lo = rp[0];
            const h8v hi = rp[1];
#pragma unroll
            for (int j = 0; j < 8; ++j) { a2[j] = lo[j]; a2[8 + j] = hi[j]; }
        }
        v8f accm = {};
        accm = __builtin_amdgcn_wmma_f32_16x16x32_f16(false, a2, false, Bw1,
                                                      (short)0, accm, false, false);

        // ---- silu + atomic scatter to sums[src] ----
        if (l16 < MSG) {
#pragma unroll
            for (int v = 0; v < 8; ++v) {
                const int m = v + lhalf * 8;
                if (m < cnt) {
                    const float msg = silu_f(accm[v] + bias1);
                    const int s = eidx[base + m];
                    atomicAdd(&sums[(size_t)s * MSG + l16], msg);
                }
            }
        }
    }
}

// ---------------------------------------------------------------------------
// Node MLP kernel: one wave computes 16 nodes per tile.
//   n_in[16 x 36 | 1 | pad->64] @ [nw0;nb0] -> silu -> hid2[16x32]
//   hid2 @ nw1[32x32] + nb1 -> silu -> silu  (outer silu from _update)
// ---------------------------------------------------------------------------
__global__ __launch_bounds__(128)
void node_mlp_kernel(const float* __restrict__ h,    // [N,32]
                     const float* __restrict__ sums, // [N,4]
                     const float* __restrict__ nw0,  // [36,32] layer slice
                     const float* __restrict__ nb0,  // [32]
                     const float* __restrict__ nw1,  // [32,32]
                     const float* __restrict__ nb1,  // [32]
                     float* __restrict__ hout,       // [N,32]
                     int nNodes)
{
    const int lane  = threadIdx.x & 31;
    const int wave  = threadIdx.x >> 5;
    const int l16   = lane & 15;
    const int lhalf = lane >> 4;

    __shared__ __align__(16) _Float16 lds[WPB][16 * 64 + 16 * 32];
    _Float16* panel = &lds[wave][0];
    _Float16* hidT  = &lds[wave][16 * 64];

    // K row 36 carries nb0 (A column 36 is the constant 1.0)
    v16h Bn0[2][2];
#pragma unroll
    for (int c = 0; c < 2; ++c)
#pragma unroll
        for (int nh = 0; nh < 2; ++nh) {
            const int n  = nh * 16 + l16;
            const int kb = c * 32 + lhalf * 16;
#pragma unroll
            for (int j = 0; j < 16; ++j) {
                const int k = kb + j;
                float w = 0.0f;
                if (k < 36)       w = nw0[k * 32 + n];
                else if (k == 36) w = nb0[n];
                Bn0[c][nh][j] = (_Float16)w;
            }
        }
    v16h Bn1[2];
#pragma unroll
    for (int nh = 0; nh < 2; ++nh) {
        const int n  = nh * 16 + l16;
        const int kb = lhalf * 16;
#pragma unroll
        for (int j = 0; j < 16; ++j)
            Bn1[nh][j] = (_Float16)nw1[(kb + j) * 32 + n];
    }
    const float b1a = nb1[l16];
    const float b1b = nb1[16 + l16];

    const int numTiles = (nNodes + 15) >> 4;
    const int gwave    = blockIdx.x * WPB + wave;
    const int wstride  = gridDim.x * WPB;

    for (int t = gwave; t < numTiles; t += wstride) {
        const int base = t << 4;
        const int cnt  = (nNodes - base < 16) ? (nNodes - base) : 16;

        const int nr = lane >> 1;
        const int eh = lane & 1;
        _Float16* row = panel + nr * 64;
        if (nr < cnt) {
            const int nd = base + nr;
            const float* hp = h + (size_t)nd * 32 + eh * 16;
#pragma unroll
            for (int j = 0; j < 16; ++j) row[eh * 16 + j] = (_Float16)hp[j];
            if (eh == 0) {
                const float* sp = sums + (size_t)nd * MSG;
#pragma unroll
                for (int j = 0; j < 4;  ++j) row[32 + j] = (_Float16)sp[j];
                row[36] = (_Float16)1.0f;   // bias selector column
#pragma unroll
                for (int j = 0; j < 11; ++j) row[37 + j] = (_Float16)0.0f;
            } else {
#pragma unroll
                for (int j = 0; j < 16; ++j) row[48 + j] = (_Float16)0.0f;
            }
        } else {
#pragma unroll
            for (int j = 0; j < 32; ++j) row[eh * 32 + j] = (_Float16)0.0f;
        }
        asm volatile("s_wait_dscnt 0" ::: "memory");

        v8f acc0 = {}; v8f acc1 = {};
#pragma unroll
        for (int c = 0; c < 2; ++c) {
            const h8v* rp = (const h8v*)(panel + l16 * 64 + c * 32 + lhalf * 16);
            const h8v lo = rp[0];
            const h8v hi = rp[1];
            v16h a;
#pragma unroll
            for (int j = 0; j < 8; ++j) { a[j] = lo[j]; a[8 + j] = hi[j]; }
            acc0 = __builtin_amdgcn_wmma_f32_16x16x32_f16(false, a, false, Bn0[c][0],
                                                          (short)0, acc0, false, false);
            acc1 = __builtin_amdgcn_wmma_f32_16x16x32_f16(false, a, false, Bn0[c][1],
                                                          (short)0, acc1, false, false);
        }
#pragma unroll
        for (int v = 0; v < 8; ++v) {
            const int m = v + lhalf * 8;
            hidT[m * 32 + l16]      = (_Float16)silu_f(acc0[v]);
            hidT[m * 32 + 16 + l16] = (_Float16)silu_f(acc1[v]);
        }
        asm volatile("s_wait_dscnt 0" ::: "memory");

        v16h a2;
        {
            const h8v* rp = (const h8v*)(hidT + l16 * 32 + lhalf * 16);
            const h8v lo = rp[0];
            const h8v hi = rp[1];
#pragma unroll
            for (int j = 0; j < 8; ++j) { a2[j] = lo[j]; a2[8 + j] = hi[j]; }
        }
        v8f o0 = {}; v8f o1 = {};
        o0 = __builtin_amdgcn_wmma_f32_16x16x32_f16(false, a2, false, Bn1[0],
                                                    (short)0, o0, false, false);
        o1 = __builtin_amdgcn_wmma_f32_16x16x32_f16(false, a2, false, Bn1[1],
                                                    (short)0, o1, false, false);
#pragma unroll
        for (int v = 0; v < 8; ++v) {
            const int m = v + lhalf * 8;
            if (m < cnt) {
                float* orow = hout + (size_t)(base + m) * 32;
                orow[l16]      = silu_f(silu_f(o0[v] + b1a));  // conv silu + outer silu
                orow[16 + l16] = silu_f(silu_f(o1[v] + b1b));
            }
        }
    }
}

// ---------------------------------------------------------------------------
// Per-graph accumulation (scatter-mean numerators + counts)
// ---------------------------------------------------------------------------
__global__ void graph_accum_kernel(const float* __restrict__ updInt,
                                   const float* __restrict__ updExt,
                                   const int*   __restrict__ batch,
                                   float* __restrict__ gInt,   // [G,32]
                                   float* __restrict__ gExt,   // [G,32]
                                   float* __restrict__ gCnt,   // [G]
                                   int nNodes)
{
    int i = blockIdx.x * blockDim.x + threadIdx.x;
    if (i >= nNodes) return;
    const int g = batch[i];
    const float* ri = updInt + (size_t)i * 32;
    const float* re = updExt + (size_t)i * 32;
#pragma unroll 8
    for (int f = 0; f < 32; ++f) atomicAdd(&gInt[g * 32 + f], ri[f]);
#pragma unroll 8
    for (int f = 0; f < 32; ++f) atomicAdd(&gExt[g * 32 + f], re[f]);
    atomicAdd(&gCnt[g], 1.0f);
}

// ---------------------------------------------------------------------------
// Readout MLP: [G,64] -> 32 -> 1 (tiny, scalar per graph)
// ---------------------------------------------------------------------------
__global__ void readout_kernel(const float* __restrict__ gInt,
                               const float* __restrict__ gExt,
                               const float* __restrict__ gCnt,
                               const float* __restrict__ gw0,  // [64,32]
                               const float* __restrict__ gb0,  // [32]
                               const float* __restrict__ gw1,  // [32,1]
                               const float* __restrict__ gb1,  // [1]
                               float* __restrict__ out)        // [G]
{
    int g = blockIdx.x * blockDim.x + threadIdx.x;
    if (g >= N_GRAPHS) return;
    const float inv = __builtin_amdgcn_rcpf(fmaxf(gCnt[g], 1.0f));
    float gin[64];
#pragma unroll 8
    for (int f = 0; f < 32; ++f) {
        gin[f]      = gInt[g * 32 + f] * inv;
        gin[32 + f] = gExt[g * 32 + f] * inv;
    }
    float osum = gb1[0];
    for (int j = 0; j < 32; ++j) {
        float t = gb0[j];
        for (int k = 0; k < 64; ++k) t += gin[k] * gw0[k * 32 + j];
        osum += silu_f(t) * gw1[j];
    }
    out[g] = silu_f(osum);
}

// ---------------------------------------------------------------------------
// Host launcher
// ---------------------------------------------------------------------------
extern "C" void kernel_launch(void* const* d_in, const int* in_sizes, int n_in,
                              void* d_out, int out_size, void* d_ws, size_t ws_size,
                              hipStream_t stream) {
    const float* x      = (const float*)d_in[0];
    const int*   iei    = (const int*)  d_in[1];
    const float* iea    = (const float*)d_in[2];
    const int*   eei    = (const int*)  d_in[3];
    const float* eea    = (const float*)d_in[4];
    const int*   batch  = (const int*)  d_in[5];
    // params flattened in sorted-key tree order:
    // external{eb0,eb1,ew0,ew1,nb0,nb1,nw0,nw1}, gb0, gb1, gw0, gw1, internal{...}
    const float* X_eb0 = (const float*)d_in[6];
    const float* X_eb1 = (const float*)d_in[7];
    const float* X_ew0 = (const float*)d_in[8];
    const float* X_ew1 = (const float*)d_in[9];
    const float* X_nb0 = (const float*)d_in[10];
    const float* X_nb1 = (const float*)d_in[11];
    const float* X_nw0 = (const float*)d_in[12];
    const float* X_nw1 = (const float*)d_in[13];
    const float* gb0   = (const float*)d_in[14];
    const float* gb1   = (const float*)d_in[15];
    const float* gw0   = (const float*)d_in[16];
    const float* gw1   = (const float*)d_in[17];
    const float* I_eb0 = (const float*)d_in[18];
    const float* I_eb1 = (const float*)d_in[19];
    const float* I_ew0 = (const float*)d_in[20];
    const float* I_ew1 = (const float*)d_in[21];
    const float* I_nb0 = (const float*)d_in[22];
    const float* I_nb1 = (const float*)d_in[23];
    const float* I_nw0 = (const float*)d_in[24];
    const float* I_nw1 = (const float*)d_in[25];

    // workspace carve-up
    float* ws     = (float*)d_ws;
    float* bufA   = ws;                                 // [N,32] ping
    float* bufInt = bufA   + (size_t)N_NODES * 32;      // [N,32] internal result
    float* bufExt = bufInt + (size_t)N_NODES * 32;      // [N,32] external result
    float* sums   = bufExt + (size_t)N_NODES * 32;      // [N,4]  edge msg sums
    float* gInt   = sums   + (size_t)N_NODES * MSG;     // [G,32]
    float* gExt   = gInt   + (size_t)N_GRAPHS * 32;     // [G,32]
    float* gCnt   = gExt   + (size_t)N_GRAPHS * 32;     // [G]

    const int EDGE_BLOCKS = 2048;   // 8192 waves over 200000 edge tiles
    const int NODE_BLOCKS = 512;    // 2048 waves over 6250 node tiles

    auto run_update = [&](const int* ei, const float* ea,
                          const float* eb0, const float* eb1,
                          const float* ew0, const float* ew1,
                          const float* nb0, const float* nb1,
                          const float* nw0, const float* nw1,
                          float* outBuf) {
        const float* hin = x;
        for (int li = 0; li < 2; ++li) {
            zero_f32<<<(N_NODES * MSG + 255) / 256, 256, 0, stream>>>(sums, N_NODES * MSG);
            edge_msg_kernel<<<EDGE_BLOCKS, 128, 0, stream>>>(
                hin, ei, ea,
                ew0 + (size_t)li * 80 * 32, eb0 + li * 32,
                ew1 + (size_t)li * 32 * MSG, eb1 + li * MSG,
                sums, N_EDGES);
            float* hout = (li == 0) ? bufA : outBuf;
            node_mlp_kernel<<<NODE_BLOCKS, 128, 0, stream>>>(
                hin, sums,
                nw0 + (size_t)li * 36 * 32, nb0 + li * 32,
                nw1 + (size_t)li * 32 * 32, nb1 + li * 32,
                hout, N_NODES);
            hin = bufA;
        }
    };

    run_update(iei, iea, I_eb0, I_eb1, I_ew0, I_ew1, I_nb0, I_nb1, I_nw0, I_nw1, bufInt);
    run_update(eei, eea, X_eb0, X_eb1, X_ew0, X_ew1, X_nb0, X_nb1, X_nw0, X_nw1, bufExt);

    // graph pooling accumulators (gInt|gExt|gCnt contiguous)
    zero_f32<<<(N_GRAPHS * 65 + 255) / 256, 256, 0, stream>>>(gInt, N_GRAPHS * 65);
    graph_accum_kernel<<<(N_NODES + 255) / 256, 256, 0, stream>>>(
        bufInt, bufExt, batch, gInt, gExt, gCnt, N_NODES);
    readout_kernel<<<1, N_GRAPHS, 0, stream>>>(
        gInt, gExt, gCnt, gw0, gb0, gw1, gb1, (float*)d_out);
}
